// HSoftmaxLayer_37417755083605
// MI455X (gfx1250) — matrix operand order repaired and checked by reading, not compiled, not run on hardware
//
#include <hip/hip_runtime.h>
#include <hip/hip_bf16.h>
#include <math.h>

typedef __attribute__((ext_vector_type(16))) _Float16 v16h;
typedef __attribute__((ext_vector_type(8)))  float    v8f;

#define D_DIM   512
#define INNER   31999            // internal Huffman nodes
#define NPAD    32000            // padded logit row stride
#define ROWS    64               // B*S
#define TDEPTH  18
#define VOCAB   32000
#define KSTEPS  (D_DIM / 32)     // 16 WMMA K-steps of 32
#define NSUB    4                // N-sub-tiles (of 16 nodes) per block
#define BTILE   (NSUB * 16)      // 64 nodes per block

// ---------------------------------------------------------------------------
// Kernel 1: z[m][n] = att[m,:] . weight[n,:]   (raw logits, no sigmoid)
// grid.x = 500 blocks of 64 nodes; block = 128 threads = 4 waves.
// Wave w computes four 16x16 tiles (m-rows [16w,16w+16) x 4 n-sub-tiles),
// reusing one A fragment per K-step across all four WMMAs.
// weight tiles staged to LDS as f16 in exact B-fragment order (64 KB dynamic).
// ---------------------------------------------------------------------------
__global__ __launch_bounds__(128)
void hs_gemm_logits(const float* __restrict__ att,
                    const float* __restrict__ weight,
                    float* __restrict__ z) {
  // B fragments: NSUB x KSTEPS x 32 lanes (v16h each) = 64 KB
  extern __shared__ v16h bfragv[];
  _Float16* bfrag = reinterpret_cast<_Float16*>(bfragv);

  const int tid = threadIdx.x;
  const int n0  = blockIdx.x * BTILE;

  // ---- stage weight tile (64 nodes x 512 k), f32 -> f16, fragment order ----
  // pair (k,k+1) of node n -> half index ((n>>4)*16 + k>>5? ) see mapping:
  //   subtile = n>>4, nloc = n&15, kstep = k>>5, hi = (k&31)>>4, e = k&15,
  //   lane = hi*16 + nloc, half idx = (subtile*KSTEPS + kstep)*512 + lane*16 + e
  for (int p = tid; p < (BTILE * D_DIM) / 2; p += 128) {
    const int el    = p * 2;
    const int n     = el >> 9;        // / 512 : 0..63
    const int k     = el & 511;
    const int st    = n >> 4;
    const int nloc  = n & 15;
    const int kstep = k >> 5;
    const int r     = k & 31;
    const int hi    = r >> 4;
    const int e     = r & 15;         // even
    const int lane  = (hi << 4) | nloc;
    float2 w2;
    if (n0 + n < INNER)
      w2 = *reinterpret_cast<const float2*>(weight + (size_t)(n0 + n) * D_DIM + k);
    else
      w2 = make_float2(0.f, 0.f);
    const unsigned lo   = __builtin_bit_cast(unsigned short, (_Float16)w2.x);
    const unsigned hi16 = __builtin_bit_cast(unsigned short, (_Float16)w2.y);
    *reinterpret_cast<unsigned*>(
        &bfrag[((st * KSTEPS + kstep) * 32 + lane) * 16 + e]) = lo | (hi16 << 16);
  }
  __syncthreads();

  const int lane = tid & 31;
  const int w    = tid >> 5;                 // wave id == m-tile id
  const int hi   = lane >> 4;
  const int mrow = w * 16 + (lane & 15);     // A-fragment row for this lane
  const float* __restrict__ arow = att + (size_t)mrow * D_DIM;

  v8f acc0 = {}, acc1 = {}, acc2 = {}, acc3 = {};
  for (int kstep = 0; kstep < KSTEPS; ++kstep) {
    // A fragment: elements 0..15 <- k0..k0+15
    const int k0 = kstep * 32 + hi * 16;
    const float4 a0 = *reinterpret_cast<const float4*>(arow + k0);
    const float4 a1 = *reinterpret_cast<const float4*>(arow + k0 + 4);
    const float4 a2 = *reinterpret_cast<const float4*>(arow + k0 + 8);
    const float4 a3 = *reinterpret_cast<const float4*>(arow + k0 + 12);
    v16h a;
    a[0]  = (_Float16)a0.x; a[1]  = (_Float16)a0.y;
    a[2]  = (_Float16)a0.z; a[3]  = (_Float16)a0.w;
    a[4]  = (_Float16)a1.x; a[5]  = (_Float16)a1.y;
    a[6]  = (_Float16)a1.z; a[7]  = (_Float16)a1.w;
    a[8]  = (_Float16)a2.x; a[9]  = (_Float16)a2.y;
    a[10] = (_Float16)a2.z; a[11] = (_Float16)a2.w;
    a[12] = (_Float16)a3.x; a[13] = (_Float16)a3.y;
    a[14] = (_Float16)a3.z; a[15] = (_Float16)a3.w;

    const v16h b0 = bfragv[(0 * KSTEPS + kstep) * 32 + lane];  // ds_load_b128 x2
    const v16h b1 = bfragv[(1 * KSTEPS + kstep) * 32 + lane];
    const v16h b2 = bfragv[(2 * KSTEPS + kstep) * 32 + lane];
    const v16h b3 = bfragv[(3 * KSTEPS + kstep) * 32 + lane];

    acc0 = __builtin_amdgcn_wmma_f32_16x16x32_f16(false, a, false, b0,
                                                  (short)0, acc0, false, false);
    acc1 = __builtin_amdgcn_wmma_f32_16x16x32_f16(false, a, false, b1,
                                                  (short)0, acc1, false, false);
    acc2 = __builtin_amdgcn_wmma_f32_16x16x32_f16(false, a, false, b2,
                                                  (short)0, acc2, false, false);
    acc3 = __builtin_amdgcn_wmma_f32_16x16x32_f16(false, a, false, b3,
                                                  (short)0, acc3, false, false);
  }

  // D layout: element r -> M = r + 8*hi, N = lane&15
  const int nloc = lane & 15;
#pragma unroll
  for (int st = 0; st < NSUB; ++st) {
    const v8f acc = (st == 0) ? acc0 : (st == 1) ? acc1 : (st == 2) ? acc2 : acc3;
    const int ncol = n0 + st * 16 + nloc;
#pragma unroll
    for (int r = 0; r < 8; ++r) {
      const int m = w * 16 + r + 8 * hi;
      z[(size_t)m * NPAD + ncol] = acc[r];
    }
  }
}

// ---------------------------------------------------------------------------
// Kernel 2: out[bs][v] = sum_t log(path prob) = sum_t -softplus(-sign*z[idx])
// One block per (b,s) row; the 125 KB logit row lives in dynamic LDS, so the
// 18 random gathers per vocab entry hit LDS. Path tables are walked at 72 B
// stride per lane -> prefetch one v-iteration ahead (global_prefetch_b8).
// ---------------------------------------------------------------------------
__global__ __launch_bounds__(256)
void hs_gather_logsum(const float* __restrict__ z,
                      const int* __restrict__ pidx,
                      const float* __restrict__ psign,
                      float* __restrict__ out) {
  extern __shared__ float s_z[];            // NPAD floats = 128000 B
  const int bs = blockIdx.x;
  const float* __restrict__ zrow = z + (size_t)bs * NPAD;

  for (int i = threadIdx.x; i < NPAD / 4; i += 256)
    reinterpret_cast<float4*>(s_z)[i] =
        reinterpret_cast<const float4*>(zrow)[i];
  __syncthreads();

  for (int v = threadIdx.x; v < VOCAB; v += 256) {
    const int base = v * TDEPTH;
    // hide latency of the next strided path-table lines
    __builtin_prefetch(pidx + base + TDEPTH * 256, 0, 1);
    __builtin_prefetch(psign + base + TDEPTH * 256, 0, 1);
    float acc = 0.f;
#pragma unroll
    for (int t = 0; t < TDEPTH; ++t) {
      const int   idx = pidx[base + t];
      const float sg  = psign[base + t];
      const float x   = -sg * s_z[idx];
      // softplus(x) = max(x,0) + log1p(exp(-|x|))  (stable)
      const float sp  = fmaxf(x, 0.f) + log1pf(__expf(-fabsf(x)));
      acc -= sp;
    }
    out[(size_t)bs * VOCAB + v] = acc;
  }
}

// ---------------------------------------------------------------------------
extern "C" void kernel_launch(void* const* d_in, const int* in_sizes, int n_in,
                              void* d_out, int out_size, void* d_ws, size_t ws_size,
                              hipStream_t stream) {
  const float* att    = (const float*)d_in[0];   // [64,512]
  const float* weight = (const float*)d_in[1];   // [31999,512]
  const int*   pidx   = (const int*)  d_in[2];   // [576000]
  const float* psign  = (const float*)d_in[3];   // [576000]
  // d_in[4] = path_bias (redundant: bias = (1-sign)/2)
  // d_in[5] = vocab_size, d_in[6] = tree_depth (compile-time constants here)

  float* z   = (float*)d_ws;                     // [64,32000] logits, 8.19 MB
  float* out = (float*)d_out;                    // [64,32000]

  const int n_blocks = (INNER + BTILE - 1) / BTILE;              // 500
  const size_t lds1  = (size_t)NSUB * KSTEPS * 32 * sizeof(v16h); // 64 KB
  hs_gemm_logits<<<n_blocks, 128, lds1, stream>>>(att, weight, z);
  hs_gather_logsum<<<ROWS, 256, NPAD * sizeof(float), stream>>>(z, pidx, psign, out);
}
